// NMS_20933670600803
// MI455X (gfx1250) — compile-verified, matrix operand order checked by reading, not run on hardware
//
#include <hip/hip_runtime.h>

typedef __attribute__((ext_vector_type(16))) _Float16 v16h;
typedef __attribute__((ext_vector_type(8)))  float    v8f;
typedef __attribute__((ext_vector_type(4)))  float    v4f;

#define NPIX 196   // 14*14

__global__ __launch_bounds__(256)
void nms_mask_kernel(const float* __restrict__ hm, float* __restrict__ out, int B) {
    const int b    = (blockIdx.x * blockDim.x + threadIdx.x) >> 5;  // wave32: one batch per wave
    const int lane = threadIdx.x & 31;
    if (b >= B) return;

    const float* __restrict__ src = hm + (size_t)b * NPIX;

    // ---- coalesced 128-bit loads: lane l holds elements [4l,4l+4) and [128+4l,128+4l+4) ----
    v4f h0 = __builtin_nontemporal_load((const v4f*)src + lane);
    v4f h1 = {0.f, 0.f, 0.f, 0.f};
    const bool hi_valid = (lane < 17);                 // 17*4 = 68 elems cover 128..195
    if (hi_valid)
        h1 = __builtin_nontemporal_load((const v4f*)(src + 128) + lane);

    float h[8];
    #pragma unroll
    for (int j = 0; j < 4; ++j) h[j]     = (h0[j] > 0.6f) ? h0[j] : 0.0f;
    #pragma unroll
    for (int j = 0; j < 4; ++j) h[4 + j] = (h1[j] > 0.6f) ? h1[j] : 0.0f;

    const int e_lo = 4 * lane;
    const int e_hi = 128 + 4 * lane;

    // ---- 4 rounds: wave-wide argmax (lowest-index tie-break) + window suppression ----
    int peak[4];
    #pragma unroll
    for (int it = 0; it < 4; ++it) {
        float bv = -1.0f;
        int   bi = 0x3fffffff;
        #pragma unroll
        for (int j = 0; j < 4; ++j)
            if (h[j] > bv) { bv = h[j]; bi = e_lo + j; }
        if (hi_valid) {
            #pragma unroll
            for (int j = 0; j < 4; ++j)
                if (h[4 + j] > bv) { bv = h[4 + j]; bi = e_hi + j; }
        }
        // butterfly reduction across the 32 lanes
        #pragma unroll
        for (int off = 16; off > 0; off >>= 1) {
            float ov = __shfl_xor(bv, off, 32);
            int   oi = __shfl_xor(bi, off, 32);
            if (ov > bv || (ov == bv && oi < bi)) { bv = ov; bi = oi; }
        }
        peak[it] = bi;

        const int x  = bi / 14, y = bi - 14 * x;
        const int x1 = max(x - 5, 0),  y1 = max(y - 5, 0);
        const int x2 = min(x + 5, 15), y2 = min(y + 5, 15);
        #pragma unroll
        for (int j = 0; j < 8; ++j) {
            const int e  = (j < 4) ? (e_lo + j) : (e_hi + j - 4);
            const int xx = e / 14, yy = e - 14 * xx;
            if ((xx >= x1) & (xx < x2) & (yy >= y1) & (yy < y2)) h[j] = 0.0f;
        }
    }

    // ---- most-distant pair among the 4 peaks (first max on ties, like jnp.argmax) ----
    int px[4], py[4];
    #pragma unroll
    for (int i = 0; i < 4; ++i) { px[i] = peak[i] / 14; py[i] = peak[i] - 14 * px[i]; }
    const int pa[6] = {0, 0, 0, 1, 1, 2};
    const int pb[6] = {1, 2, 3, 2, 3, 3};
    int bestd = -1, bp = 0;
    #pragma unroll
    for (int p = 0; p < 6; ++p) {
        const int dx = px[pb[p]] - px[pa[p]];
        const int dy = py[pb[p]] - py[pa[p]];
        const int d  = dx * dx + dy * dy;
        if (d > bestd) { bestd = d; bp = p; }
    }
    const int x0 = px[pa[bp]], y0 = py[pa[bp]];
    const int x1c = px[pb[bp]], y1c = py[pb[bp]];

    // diff(u,v) = d1 - d2 = Au*u + Bv*v + Cc  (integers, |.| <= 728 -> exact in f16/f32)
    const int Au = 2 * (x1c - x0);
    const int Bv = 2 * (y1c - y0);
    const int Cc = (x0 * x0 + y0 * y0) - (x1c * x1c + y1c * y1c);

    // ---- one WMMA evaluates the separating plane over the whole 16x16 grid ----
    // A (16x32 f16): row u = [u, 1, 0, ...];  lanes 0-15 hold K=0..7 pairs (row = lane)
    // B (32x16 f16): col v = [Au, Bv*v + Cc, 0, ...]; lanes 0-15 hold K=0..1 (col = lane)
    const bool lo = (lane < 16);
    const int  cl = lane & 15;
    v16h A, Bm;
    #pragma unroll
    for (int i = 0; i < 16; ++i) { A[i] = (_Float16)0.f; Bm[i] = (_Float16)0.f; }
    A[0]  = lo ? (_Float16)(float)cl : (_Float16)0.f;
    A[1]  = lo ? (_Float16)1.f       : (_Float16)0.f;
    Bm[0] = lo ? (_Float16)(float)Au : (_Float16)0.f;
    Bm[1] = lo ? (_Float16)(float)(Bv * cl + Cc) : (_Float16)0.f;
    v8f Cm;
    #pragma unroll
    for (int i = 0; i < 8; ++i) Cm[i] = 0.f;

    v8f D = __builtin_amdgcn_wmma_f32_16x16x32_f16(
        /*neg_a=*/false, A, /*neg_b=*/false, Bm,
        /*c_mod=*/(short)0, Cm, /*reuse_a=*/false, /*reuse_b=*/false);

    // ---- emit masks from the D layout: VGPR r -> row (lo ? r : r+8), col = lane&15 ----
    const int rbase = lo ? 0 : 8;
    float* __restrict__ o1 = out + (size_t)b * NPIX;
    float* __restrict__ o2 = out + (size_t)B * NPIX + (size_t)b * NPIX;
    #pragma unroll
    for (int r = 0; r < 8; ++r) {
        const int row = rbase + r;
        if ((row < 14) & (cl < 14)) {
            const float m1 = (D[r] < 0.0f) ? 1.0f : 0.0f;
            __builtin_nontemporal_store(m1,        &o1[row * 14 + cl]);
            __builtin_nontemporal_store(1.0f - m1, &o2[row * 14 + cl]);
        }
    }
}

extern "C" void kernel_launch(void* const* d_in, const int* in_sizes, int n_in,
                              void* d_out, int out_size, void* d_ws, size_t ws_size,
                              hipStream_t stream) {
    const float* hm  = (const float*)d_in[0];
    float*       out = (float*)d_out;
    const int B = in_sizes[0] / NPIX;          // 131072 for the reference setup

    const int threads = 256;                   // 8 wave32 per block
    const int waves_per_block = threads / 32;
    const int blocks = (B + waves_per_block - 1) / waves_per_block;
    nms_mask_kernel<<<blocks, threads, 0, stream>>>(hm, out, B);
}
